// MoeMLPDebug_21483426414712
// MI455X (gfx1250) — matrix operand order repaired and checked by reading, not compiled, not run on hardware
//
#include <hip/hip_runtime.h>
#include <stdint.h>

// MoE MLP for MI455X (gfx1250, wave32, WMMA + async-to-LDS).
// T=8192, D=1024, E=8, K=2, DFF=2048.
// One-time: cast X->bf16. Per expert: transpose-cast W1e/W2e -> bf16 N-major,
// then H = gelu(Xb @ W1e) and OUT (+)= g_e * (H @ W2e) with bf16 WMMA.
// GEMM k-loop has NO conversion and NO ds stores: both A and B tiles stream
// memory->LDS via global_load_async_to_lds_b128 (ASYNCcnt), double-buffered.

typedef __attribute__((ext_vector_type(16))) __bf16 v16bf;
typedef __attribute__((ext_vector_type(8)))  float  v8f;

#define T_TOK 8192
#define DIM   1024
#define NEXP  8
#define DFF   2048
#define W1COLS (DFF * NEXP)   // 16384
#define TILE_STRIDE 40        // ushorts; 80B rows: 16B-aligned, banks distinct

__device__ __forceinline__ unsigned short f32_to_bf16(float f) {
  unsigned u = __float_as_uint(f);
  u += 0x7FFFu + ((u >> 16) & 1u);   // round-to-nearest-even
  return (unsigned short)(u >> 16);
}

union FragBF { unsigned u[8]; v16bf v; };

// ------------------------------------------------------------ cast X->bf16 --
__global__ __launch_bounds__(256)
void cast_bf16_kernel(const float* __restrict__ X, unsigned short* __restrict__ Xb) {
  const long base = ((long)blockIdx.x * 256 + threadIdx.x) * 4;
  #pragma unroll
  for (int r = 0; r < 8; ++r) {
    const long i = base + (long)r * 1024 * 1024;
    const float4 v = *(const float4*)&X[i];
    uint2 p;
    p.x = (unsigned)f32_to_bf16(v.x) | ((unsigned)f32_to_bf16(v.y) << 16);
    p.y = (unsigned)f32_to_bf16(v.z) | ((unsigned)f32_to_bf16(v.w) << 16);
    *(uint2*)&Xb[i] = p;
  }
}

// ------------------------------------------- transpose-cast weight slice ----
// dst[n * dst_stride + k] = bf16(src[(row0 + k) * src_stride + col0 + n])
// grid.x over k (64/block), grid.y over n (64/block).
__global__ __launch_bounds__(256)
void transpose_cast_kernel(const float* __restrict__ src, long src_stride,
                           long row0, long col0,
                           unsigned short* __restrict__ dst, long dst_stride) {
  __shared__ unsigned short lt[64][72];   // [n][k], 144B rows (16B aligned)
  const int k0 = blockIdx.x * 64;
  const int n0 = blockIdx.y * 64;
  const int tid = threadIdx.x;

  #pragma unroll
  for (int r = 0; r < 4; ++r) {           // read 64(k) x 64(n) fp32, coalesced
    const int c = tid + r * 256;
    const int row = c >> 4, c4 = (c & 15) * 4;
    const float4 v = *(const float4*)&src[(row0 + k0 + row) * src_stride + col0 + n0 + c4];
    lt[c4 + 0][row] = f32_to_bf16(v.x);
    lt[c4 + 1][row] = f32_to_bf16(v.y);
    lt[c4 + 2][row] = f32_to_bf16(v.z);
    lt[c4 + 3][row] = f32_to_bf16(v.w);
  }
  __syncthreads();
  #pragma unroll
  for (int r = 0; r < 2; ++r) {           // write 64(n) x 64(k) bf16, b128
    const int c = tid + r * 256;
    const int row = c >> 3, cc = (c & 7) * 8;
    const uint4 p = *(const uint4*)&lt[row][cc];
    *(uint4*)&dst[(long)(n0 + row) * dst_stride + k0 + cc] = p;
  }
}

// ---------------------------------------------------------------- router ----
__global__ __launch_bounds__(256)
void router_kernel(const float* __restrict__ x,
                   const float* __restrict__ rw,
                   float* __restrict__ gates) {
  const int token = blockIdx.x * 8 + (threadIdx.x >> 5);
  const int lane  = threadIdx.x & 31;
  const float* xr = x + (long)token * DIM;

  float acc[NEXP];
  #pragma unroll
  for (int e = 0; e < NEXP; ++e) acc[e] = 0.f;

  for (int d = lane; d < DIM; d += 32) {
    const float xv = xr[d];
    #pragma unroll
    for (int e = 0; e < NEXP; ++e) acc[e] += xv * rw[d * NEXP + e];
  }
  #pragma unroll
  for (int off = 16; off >= 1; off >>= 1) {
    #pragma unroll
    for (int e = 0; e < NEXP; ++e) acc[e] += __shfl_xor(acc[e], off, 32);
  }
  if (lane == 0) {
    float mx = acc[0];
    #pragma unroll
    for (int e = 1; e < NEXP; ++e) mx = fmaxf(mx, acc[e]);
    float p[NEXP], den = 0.f;
    #pragma unroll
    for (int e = 0; e < NEXP; ++e) { p[e] = __expf(acc[e] - mx); den += p[e]; }
    const float inv = 1.f / den;
    #pragma unroll
    for (int e = 0; e < NEXP; ++e) p[e] *= inv;
    int i1 = 0;
    #pragma unroll
    for (int e = 1; e < NEXP; ++e) if (p[e] > p[i1]) i1 = e;
    int i2 = (i1 == 0) ? 1 : 0;
    #pragma unroll
    for (int e = 0; e < NEXP; ++e) if (e != i1 && p[e] > p[i2]) i2 = e;
    const float s = 1.f / (p[i1] + p[i2]);
    #pragma unroll
    for (int e = 0; e < NEXP; ++e) gates[(long)token * NEXP + e] = 0.f;
    gates[(long)token * NEXP + i1] = p[i1] * s;
    gates[(long)token * NEXP + i2] = p[i2] * s;
  }
}

// ----------------------------------------------------------- WMMA helpers ---
// A frag 16x32 bf16: lane l(0..15)=row M=l; k-pairs {0,2,4,6} and {16,18,20,22};
// lanes 16..31: K+8 -> two 16B contiguous ds_load_b128.
// B frag 32x16 bf16 (N-major LDS): lane n = col; same K order as A frag:
// lanes 0..15 K=0..7,16..23; lanes 16..31 K=8..15,24..31.  (ISA 7.12.2)
__device__ __forceinline__ void load_a_frag(FragBF& f, const unsigned short* la,
                                            int mrow, int l, int hi) {
  const int kb[8] = {0, 2, 4, 6, 16, 18, 20, 22};
  const unsigned short* base = la + (mrow + l) * TILE_STRIDE + hi * 8;
  #pragma unroll
  for (int i = 0; i < 8; ++i)
    f.u[i] = *(const unsigned*)(base + kb[i]);
}

__device__ __forceinline__ void load_b_frag(FragBF& f, const unsigned short* lbt,
                                            int ncol, int l, int hi) {
  const unsigned short* base = lbt + (ncol + l) * TILE_STRIDE + hi * 8;
  #pragma unroll
  for (int i = 0; i < 8; ++i)
    f.u[i] = *(const unsigned*)(base + 2 * i + (i >= 4 ? 8 : 0));
}

// Async raw copy of a 128x32 bf16 tile (memory -> LDS), 2 x B128 per thread.
__device__ __forceinline__ void async_tile(const unsigned short* __restrict__ src,
                                           long row_stride, int r0, int kk,
                                           unsigned short* dst, int tid) {
  #pragma unroll
  for (int r = 0; r < 2; ++r) {
    const int c = tid + r * 256;          // 512 chunks of 8 bf16
    const int row = c >> 2, cc = c & 3;   // 4 chunks per 32-elem row
    const unsigned short* g = src + (long)(r0 + row) * row_stride + kk + cc * 8;
    const unsigned lds = (unsigned)(uintptr_t)(dst + row * TILE_STRIDE + cc * 8);
    asm volatile("global_load_async_to_lds_b128 %0, %1, off"
                 :: "v"(lds), "v"((unsigned long long)(uintptr_t)g)
                 : "memory");
  }
}

__device__ __forceinline__ void wait_async_and_barrier() {
  asm volatile("s_wait_asynccnt 0" ::: "memory");
  __syncthreads();
}

// ------------------------------------------------------------- GEMM core ----
// C_tile = A[M=128, K] @ Bt[N=128, K]^T with bf16 WMMA, double-buffered LDS.
// Wave tile 32x64 (8 WMMA / 32-deep k-step). Accumulators left in acc[2][4].
template <int NK>
__device__ __forceinline__ void gemm_core(const unsigned short* __restrict__ A,
                                          long a_stride,
                                          const unsigned short* __restrict__ Bt,
                                          long b_stride,
                                          int m0, int n0, int tid,
                                          unsigned short (*la)[128 * TILE_STRIDE],
                                          unsigned short (*lb)[128 * TILE_STRIDE],
                                          int mrow, int nw, int l, int hi,
                                          v8f acc[2][4]) {
  async_tile(A,  a_stride, m0, 0, la[0], tid);
  async_tile(Bt, b_stride, n0, 0, lb[0], tid);
  wait_async_and_barrier();

  for (int s = 0; s < NK; ++s) {
    const int cur = s & 1, nxt = cur ^ 1;
    if (s + 1 < NK) {                    // stage next k-step (overlaps compute)
      const int kk = (s + 1) * 32;
      async_tile(A,  a_stride, m0, kk, la[nxt], tid);
      async_tile(Bt, b_stride, n0, kk, lb[nxt], tid);
    }

    FragBF af[2], bf[4];
    load_a_frag(af[0], la[cur], mrow,      l, hi);
    load_a_frag(af[1], la[cur], mrow + 16, l, hi);
    #pragma unroll
    for (int t = 0; t < 4; ++t)
      load_b_frag(bf[t], lb[cur], nw * 64 + t * 16, l, hi);
    #pragma unroll
    for (int a = 0; a < 2; ++a)
      #pragma unroll
      for (int t = 0; t < 4; ++t)
        acc[a][t] = __builtin_amdgcn_wmma_f32_16x16x32_bf16(
            false, af[a].v, false, bf[t].v, (short)0, acc[a][t], false, false);

    wait_async_and_barrier();
  }
}

// ------------------------------------------------------------------ GEMM1 ---
// H[T, DFF] = gelu(Xb[T, D] @ W1tb[DFF, D]^T) stored as bf16 bits.
__global__ __launch_bounds__(256)
void gemm1_gelu(const unsigned short* __restrict__ Xb,
                const unsigned short* __restrict__ W1tb,
                unsigned short* __restrict__ H) {
  __shared__ unsigned short la[2][128 * TILE_STRIDE];
  __shared__ unsigned short lb[2][128 * TILE_STRIDE];

  const int m0 = blockIdx.x * 128;
  const int n0 = blockIdx.y * 128;
  const int tid = threadIdx.x;
  const int wave = tid >> 5, lane = tid & 31;
  const int mrow = (wave & 3) * 32, nw = wave >> 2;
  const int l = lane & 15, hi = lane >> 4;

  v8f acc[2][4];
  #pragma unroll
  for (int a = 0; a < 2; ++a)
    #pragma unroll
    for (int t = 0; t < 4; ++t)
      #pragma unroll
      for (int i = 0; i < 8; ++i) acc[a][t][i] = 0.f;

  gemm_core<DIM / 32>(Xb, DIM, W1tb, DIM, m0, n0, tid, la, lb, mrow, nw, l, hi, acc);

  #pragma unroll
  for (int a = 0; a < 2; ++a)
    #pragma unroll
    for (int t = 0; t < 4; ++t) {
      const int ng = n0 + nw * 64 + t * 16 + l;
      #pragma unroll
      for (int i = 0; i < 8; ++i) {
        const int mg = m0 + mrow + a * 16 + hi * 8 + i;
        const float v = acc[a][t][i];
        const float g = 0.5f * v * (1.0f + erff(v * 0.70710678118654752f));
        H[(long)mg * DFF + ng] = f32_to_bf16(g);
      }
    }
}

// ------------------------------------------------------------------ GEMM2 ---
// OUT[T, D] (+)= gate_e(m) * (H[T, DFF] @ W2tb[D, DFF]^T)
__global__ __launch_bounds__(256)
void gemm2_scatter(const unsigned short* __restrict__ H,
                   const unsigned short* __restrict__ W2tb,
                   const float* __restrict__ gates,
                   float* __restrict__ OUT, int expert, int beta) {
  __shared__ unsigned short la[2][128 * TILE_STRIDE];
  __shared__ unsigned short lb[2][128 * TILE_STRIDE];

  const int m0 = blockIdx.x * 128;
  const int n0 = blockIdx.y * 128;
  const int tid = threadIdx.x;
  const int wave = tid >> 5, lane = tid & 31;
  const int mrow = (wave & 3) * 32, nw = wave >> 2;
  const int l = lane & 15, hi = lane >> 4;

  v8f acc[2][4];
  #pragma unroll
  for (int a = 0; a < 2; ++a)
    #pragma unroll
    for (int t = 0; t < 4; ++t)
      #pragma unroll
      for (int i = 0; i < 8; ++i) acc[a][t][i] = 0.f;

  gemm_core<DFF / 32>(H, DFF, W2tb, DFF, m0, n0, tid, la, lb, mrow, nw, l, hi, acc);

  #pragma unroll
  for (int a = 0; a < 2; ++a)
    #pragma unroll
    for (int t = 0; t < 4; ++t) {
      const int ng = n0 + nw * 64 + t * 16 + l;
      #pragma unroll
      for (int i = 0; i < 8; ++i) {
        const int mg = m0 + mrow + a * 16 + hi * 8 + i;
        const float g = gates[(long)mg * NEXP + expert];
        const float v = g * acc[a][t][i];
        const long o = (long)mg * DIM + ng;
        OUT[o] = beta ? (OUT[o] + v) : v;
      }
    }
}

// ------------------------------------------------------------------- host ---
extern "C" void kernel_launch(void* const* d_in, const int* in_sizes, int n_in,
                              void* d_out, int out_size, void* d_ws, size_t ws_size,
                              hipStream_t stream) {
  const float* x        = (const float*)d_in[0];   // [T, D]
  const float* router_w = (const float*)d_in[1];   // [D, E]
  const float* w1       = (const float*)d_in[2];   // [D, DFF*E]
  const float* w2       = (const float*)d_in[3];   // [DFF*E, D]
  float* out = (float*)d_out;                      // [T, D]

  // ws: gates | Xb [T*D bf16] | H [T*DFF bf16] | W1tb [DFF*D bf16] | W2tb [D*DFF bf16]
  char* ws = (char*)d_ws;
  float* gates = (float*)ws;
  size_t off = ((size_t)T_TOK * NEXP * sizeof(float) + 255) & ~(size_t)255;
  unsigned short* Xb = (unsigned short*)(ws + off);
  off += (size_t)T_TOK * DIM * 2;
  unsigned short* H = (unsigned short*)(ws + off);
  off += (size_t)T_TOK * DFF * 2;
  unsigned short* W1tb = (unsigned short*)(ws + off);   // [n<DFF][k<D]
  off += (size_t)DFF * DIM * 2;
  unsigned short* W2tb = (unsigned short*)(ws + off);   // [n<D][k<DFF]

  cast_bf16_kernel<<<1024, 256, 0, stream>>>(x, Xb);
  router_kernel<<<T_TOK / 8, 256, 0, stream>>>(x, router_w, gates);

  for (int e = 0; e < NEXP; ++e) {
    // W1e: src rows k<D, cols (e*DFF + n), n<DFF  ->  W1tb[n][k]
    transpose_cast_kernel<<<dim3(DIM / 64, DFF / 64), 256, 0, stream>>>(
        w1, W1COLS, 0, (long)e * DFF, W1tb, DIM);
    // W2e: src rows (e*DFF + k), k<DFF, cols n<D  ->  W2tb[n][k]
    transpose_cast_kernel<<<dim3(DFF / 64, DIM / 64), 256, 0, stream>>>(
        w2, DIM, (long)e * DFF, 0, W2tb, DFF);

    gemm1_gelu<<<dim3(T_TOK / 128, DFF / 128), 256, 0, stream>>>(Xb, W1tb, H);
    gemm2_scatter<<<dim3(T_TOK / 128, DIM / 128), 256, 0, stream>>>(H, W2tb, gates,
                                                                    out, e, e != 0);
  }
}
